// VJEPA_1314259992840
// MI455X (gfx1250) — compile-verified, matrix operand order checked by reading, not run on hardware
//
#include <hip/hip_runtime.h>
#include <hip/hip_bf16.h>
#include <math.h>

// ---------------- problem constants ----------------
#define D_    768
#define H_    12
#define HD_   64          // head dim
#define DFF_  3072
#define DEPTH_ 6
#define NTOK  196
#define GRIDP 14
#define PSZ   16
#define BATCH 64
#define PREDD 384
#define MTOK  (BATCH * NTOK)      // 12544
#define QKVD  (3 * D_)            // 2304
#define QKV_PADROWS 64            // zeroed tail rows after MTOK
#define NPAD  224                 // padded key length for WMMA (7*32)
#define OUTEL (MTOK * D_)         // 9,633,792 per tensor

// GEMM blocking: block tile 128x128, 8 waves in 4(M) x 2(N), wave tile 32x64
#define BM 128
#define BN 128
#define BK 32

typedef __attribute__((ext_vector_type(8)))  __bf16 v8bf;
typedef __attribute__((ext_vector_type(16))) __bf16 v16bf;
typedef __attribute__((ext_vector_type(8)))  float  v8f;

// ---------------- device helpers ----------------
__device__ inline v8f wmma_bf16(v16bf a, v16bf b, v8f c) {
  return __builtin_amdgcn_wmma_f32_16x16x32_bf16(
      /*neg_a=*/false, a, /*neg_b=*/false, b,
      /*c_mod=*/(short)0, c, /*reuse_a=*/false, /*reuse_b=*/false);
}

// A/B fragment, 16-bit elements, rows contiguous with stride ldk (generic ptr:
// works for global or LDS). ISA 7.12.2: lane%16 -> row, lane/16 -> K group;
// element j: K(j) = (j<8?0:16) + (lane>>4)*8 + (j&7) -> two 16B vector loads.
__device__ inline v16bf load_frag_rows(const __bf16* base, int ldk, int rowBase,
                                       int k0, int lane) {
  int r  = lane & 15;
  int kg = lane >> 4;
  const __bf16* p = base + (size_t)(rowBase + r) * ldk + k0 + kg * 8;
  v8bf lo = *(const v8bf*)(p);
  v8bf hi = *(const v8bf*)(p + 16);
  return __builtin_shufflevector(lo, hi, 0,1,2,3,4,5,6,7,8,9,10,11,12,13,14,15);
}

// 8 WMMAs against one staged 128x32 A-slab / 128x32 B-slab
__device__ inline void compute_slab(const __bf16* As, const __bf16* Bs, int mWave,
                                    int nWave, int lane, v8f acc[2][4]) {
  v16bf af[2], bf[4];
#pragma unroll
  for (int mi = 0; mi < 2; ++mi)
    af[mi] = load_frag_rows(As, BK, mWave + mi * 16, 0, lane);
#pragma unroll
  for (int ni = 0; ni < 4; ++ni)
    bf[ni] = load_frag_rows(Bs, BK, nWave + ni * 16, 0, lane);
#pragma unroll
  for (int mi = 0; mi < 2; ++mi)
#pragma unroll
    for (int ni = 0; ni < 4; ++ni)
      acc[mi][ni] = wmma_bf16(af[mi], bf[ni], acc[mi][ni]);
}

// ---------------- elementwise / utility kernels ----------------
__global__ void cvt_f32_bf16_kernel(const float* __restrict__ src,
                                    __bf16* __restrict__ dst, size_t n) {
  for (size_t i = blockIdx.x * (size_t)blockDim.x + threadIdx.x; i < n;
       i += (size_t)gridDim.x * blockDim.x)
    dst[i] = (__bf16)src[i];
}

__global__ void zero_bf16_kernel(__bf16* p, size_t n) {
  for (size_t i = blockIdx.x * (size_t)blockDim.x + threadIdx.x; i < n;
       i += (size_t)gridDim.x * blockDim.x)
    p[i] = (__bf16)0.0f;
}

__global__ void zero_f32_kernel(float* p, size_t n) {
  for (size_t i = blockIdx.x * (size_t)blockDim.x + threadIdx.x; i < n;
       i += (size_t)gridDim.x * blockDim.x)
    p[i] = 0.0f;
}

__global__ void copy_f32_kernel(const float* __restrict__ s, float* __restrict__ d,
                                size_t n) {
  for (size_t i = blockIdx.x * (size_t)blockDim.x + threadIdx.x; i < n;
       i += (size_t)gridDim.x * blockDim.x)
    d[i] = s[i];
}

// imgs [B,3,224,224] -> patches bf16 [MTOK, 768] with col = c*256 + py*16 + px
__global__ void patch_extract_kernel(const float* __restrict__ imgs,
                                     __bf16* __restrict__ out) {
  size_t total = (size_t)MTOK * D_;
  for (size_t i = blockIdx.x * (size_t)blockDim.x + threadIdx.x; i < total;
       i += (size_t)gridDim.x * blockDim.x) {
    int row = (int)(i / D_);
    int col = (int)(i % D_);
    int b = row / NTOK, n = row % NTOK;
    int gh = n / GRIDP, gw = n % GRIDP;
    int c = col / (PSZ * PSZ), rc = col % (PSZ * PSZ);
    int py = rc / PSZ, px = rc % PSZ;
    size_t src = (((size_t)b * 3 + c) * (GRIDP * PSZ) + gh * PSZ + py) *
                     (GRIDP * PSZ) + gw * PSZ + px;
    out[i] = (__bf16)imgs[src];
  }
}

__global__ void add_pos_kernel(float* __restrict__ x, const float* __restrict__ pos) {
  size_t total = (size_t)MTOK * D_;
  for (size_t i = blockIdx.x * (size_t)blockDim.x + threadIdx.x; i < total;
       i += (size_t)gridDim.x * blockDim.x)
    x[i] += pos[i % ((size_t)NTOK * D_)];
}

// stable compaction of kept tokens to front, zero pad tail (per batch)
__global__ void gather_pad_kernel(const float* __restrict__ x,
                                  const unsigned char* __restrict__ mask, int invert,
                                  float* __restrict__ out) {
  int b = blockIdx.x;
  __shared__ int order[NTOK];
  if (threadIdx.x == 0) {
    int c = 0;
    for (int n = 0; n < NTOK; ++n) {
      int keep = mask[b * NTOK + n] ? 1 : 0;
      if (invert) keep = !keep;
      if (keep) order[c++] = n;
    }
    for (int i = c; i < NTOK; ++i) order[i] = -1;
  }
  __syncthreads();
  for (int i = threadIdx.x; i < NTOK * D_; i += blockDim.x) {
    int tok = i / D_, d = i % D_;
    int src = order[tok];
    out[(size_t)b * NTOK * D_ + i] =
        (src >= 0) ? x[((size_t)b * NTOK + src) * D_ + d] : 0.0f;
  }
}

// per-token LayerNorm -> bf16
__global__ void ln_kernel(const float* __restrict__ x, const float* __restrict__ g,
                          const float* __restrict__ bta, __bf16* __restrict__ out) {
  int row = blockIdx.x;
  const float* xr = x + (size_t)row * D_;
  __shared__ float rs[256], rs2[256];
  float s = 0.f, s2 = 0.f;
  for (int i = threadIdx.x; i < D_; i += 256) {
    float v = xr[i];
    s += v; s2 += v * v;
  }
  rs[threadIdx.x] = s; rs2[threadIdx.x] = s2;
  __syncthreads();
  for (int o = 128; o > 0; o >>= 1) {
    if (threadIdx.x < o) {
      rs[threadIdx.x] += rs[threadIdx.x + o];
      rs2[threadIdx.x] += rs2[threadIdx.x + o];
    }
    __syncthreads();
  }
  float mu  = rs[0] * (1.0f / D_);
  float var = rs2[0] * (1.0f / D_) - mu * mu;
  float inv = rsqrtf(var + 1e-5f);
  for (int i = threadIdx.x; i < D_; i += 256)
    out[(size_t)row * D_ + i] = (__bf16)((xr[i] - mu) * inv * g[i] + bta[i]);
}

// ------- WMMA GEMM: C[M,N] = act(A[M,K] * W[N,K]^T + bias) (+residual) -------
// Double-buffered LDS staging; 128x128 block tile; 8 waves (4M x 2N); wave tile
// 32x64 (2 M-frags x 4 N-frags = 8 f32 accumulators). K/32 must be even.
// act: 0 none, 1 relu, 2 gelu(exact)
__global__ void __launch_bounds__(256)
gemm_bf16_kernel(const __bf16* __restrict__ A, const __bf16* __restrict__ W,
                 const float* __restrict__ bias, const float* __restrict__ residual,
                 float* __restrict__ Cf, __bf16* __restrict__ Cbf,
                 int M, int Nn, int K, int act) {
  __shared__ __align__(16) __bf16 As0[BM * BK], As1[BM * BK];   // 8 KB each
  __shared__ __align__(16) __bf16 Bs0[BN * BK], Bs1[BN * BK];

  int tid  = threadIdx.x;
  int lane = tid & 31;
  int wave = tid >> 5;
  int mWave = (wave >> 1) * 32;          // row offset inside block tile
  int nWave = (wave & 1) * 64;           // col offset inside block tile
  int blockMRow = blockIdx.y * BM;
  int blockNRow = blockIdx.x * BN;

  // cooperative-load chunk coords: 4 x 16B chunks per 32-elem row; 512 chunks,
  // each thread owns chunks tid and tid+256.
  int c0row = tid >> 2, c0off = (tid & 3) * 8;
  int c1row = (tid + 256) >> 2, c1off = ((tid + 256) & 3) * 8;
  int sOff0 = c0row * BK + c0off;
  int sOff1 = c1row * BK + c1off;
  const __bf16* pA0 = A + (size_t)(blockMRow + c0row) * K + c0off;
  const __bf16* pA1 = A + (size_t)(blockMRow + c1row) * K + c1off;
  const __bf16* pW0 = W + (size_t)(blockNRow + c0row) * K + c0off;
  const __bf16* pW1 = W + (size_t)(blockNRow + c1row) * K + c1off;

  v8f acc[2][4];
#pragma unroll
  for (int mi = 0; mi < 2; ++mi)
#pragma unroll
    for (int ni = 0; ni < 4; ++ni) acc[mi][ni] = (v8f){};

  // preload slab 0 into buf0 (distinct temporaries -> clause of 4 loads)
  v8bf ra0 = *(const v8bf*)(pA0);
  v8bf ra1 = *(const v8bf*)(pA1);
  v8bf rb0 = *(const v8bf*)(pW0);
  v8bf rb1 = *(const v8bf*)(pW1);
  *(v8bf*)(As0 + sOff0) = ra0;
  *(v8bf*)(As0 + sOff1) = ra1;
  *(v8bf*)(Bs0 + sOff0) = rb0;
  *(v8bf*)(Bs0 + sOff1) = rb1;
  __syncthreads();

  int nk = K / BK;   // even for all K used here (24, 96, 12)
  for (int k = 0; k < nk; k += 2) {
    // issue global loads for slab k+1 (always exists), overlap with compute
    ra0 = *(const v8bf*)(pA0 + (size_t)(k + 1) * BK);
    ra1 = *(const v8bf*)(pA1 + (size_t)(k + 1) * BK);
    rb0 = *(const v8bf*)(pW0 + (size_t)(k + 1) * BK);
    rb1 = *(const v8bf*)(pW1 + (size_t)(k + 1) * BK);
    compute_slab(As0, Bs0, mWave, nWave, lane, acc);
    *(v8bf*)(As1 + sOff0) = ra0;
    *(v8bf*)(As1 + sOff1) = ra1;
    *(v8bf*)(Bs1 + sOff0) = rb0;
    *(v8bf*)(Bs1 + sOff1) = rb1;
    __syncthreads();

    bool more = (k + 2) < nk;
    if (more) {
      ra0 = *(const v8bf*)(pA0 + (size_t)(k + 2) * BK);
      ra1 = *(const v8bf*)(pA1 + (size_t)(k + 2) * BK);
      rb0 = *(const v8bf*)(pW0 + (size_t)(k + 2) * BK);
      rb1 = *(const v8bf*)(pW1 + (size_t)(k + 2) * BK);
      __builtin_prefetch(pA0 + (size_t)(k + 3) * BK, 0, 1);  // global_prefetch_b8
      __builtin_prefetch(pW0 + (size_t)(k + 3) * BK, 0, 1);
    }
    compute_slab(As1, Bs1, mWave, nWave, lane, acc);
    if (more) {
      *(v8bf*)(As0 + sOff0) = ra0;
      *(v8bf*)(As0 + sOff1) = ra1;
      *(v8bf*)(Bs0 + sOff0) = rb0;
      *(v8bf*)(Bs0 + sOff1) = rb1;
    }
    __syncthreads();
  }

  // epilogue
  int mh = 8 * (lane >> 4);
#pragma unroll
  for (int mi = 0; mi < 2; ++mi) {
#pragma unroll
    for (int ni = 0; ni < 4; ++ni) {
      int n = blockNRow + nWave + ni * 16 + (lane & 15);
      float bv = bias ? bias[n] : 0.0f;
#pragma unroll
      for (int r = 0; r < 8; ++r) {
        int m = blockMRow + mWave + mi * 16 + mh + r;
        float v = acc[mi][ni][r] + bv;
        if (act == 1)      v = fmaxf(v, 0.0f);
        else if (act == 2) v = 0.5f * v * (1.0f + erff(v * 0.70710678118654752f));
        if (residual) v += residual[(size_t)m * Nn + n];
        if (Cf)  Cf[(size_t)m * Nn + n]  = v;
        if (Cbf) Cbf[(size_t)m * Nn + n] = (__bf16)v;
      }
    }
  }
}

// ---------------- attention: one block per (qtile, head, batch) -----------------
// qkv bf16 [MTOK + QKV_PADROWS, 2304]; out bf16 [MTOK, 768]
__global__ void __launch_bounds__(256)
attn_kernel(const __bf16* __restrict__ qkv, __bf16* __restrict__ attn_out) {
  int qt = blockIdx.x;        // 0..12 (13 tiles of 16 queries, last partial)
  int h  = blockIdx.y;        // 0..11
  int b  = blockIdx.z;        // 0..63
  int lane = threadIdx.x & 31;
  int wave = threadIdx.x >> 5;

  __shared__ __align__(16) float  sc[16 * NPAD];   // scores (f32)
  __shared__ __align__(16) __bf16 pr[16 * NPAD];   // probs (bf16)
  __shared__ __align__(16) __bf16 vt[HD_ * NPAD];  // V^T: [headdim][token]

  const int ld = QKVD;
  const __bf16* qb = qkv + (size_t)(b * NTOK) * ld + h * HD_;            // Q cols
  const __bf16* kb = qkv + (size_t)(b * NTOK) * ld + D_ + h * HD_;       // K cols
  const __bf16* vb = qkv + (size_t)(b * NTOK) * ld + 2 * D_ + h * HD_;   // V cols

  // stage V^T into LDS: coalesced reads (consecutive lanes -> consecutive dims)
  for (int i = threadIdx.x; i < HD_ * NPAD; i += 256) {
    int k = i >> 6;        // token 0..223 (tail rows are zero / masked by probs)
    int n = i & (HD_ - 1); // head dim 0..63
    vt[n * NPAD + k] = vb[(size_t)k * ld + n];
  }

  // hoist Q fragments (16x64 -> two k-steps of 32)
  v16bf aq[2];
#pragma unroll
  for (int kk = 0; kk < 2; ++kk)
    aq[kk] = load_frag_rows(qb, ld, qt * 16, kk * 32, lane);

  // scores: S[16 x 208] = Q(16x64) * K^T, 13 key tiles spread over 8 waves
  for (int kt = wave; kt < 13; kt += 8) {
    v8f acc = {};
#pragma unroll
    for (int kk = 0; kk < 2; ++kk) {
      v16bf bb = load_frag_rows(kb, ld, kt * 16, kk * 32, lane);
      acc = wmma_bf16(aq[kk], bb, acc);
    }
    int n  = lane & 15;
    int mh = 8 * (lane >> 4);
#pragma unroll
    for (int r = 0; r < 8; ++r)
      sc[(mh + r) * NPAD + kt * 16 + n] = acc[r] * 0.125f;  // 1/sqrt(64)
  }
  __syncthreads();

  // softmax over 196 keys per query row; bf16 probs, zero pad columns
  if (threadIdx.x < 16) {
    float* row = sc + threadIdx.x * NPAD;
    float mx = -1e30f;
    for (int c = 0; c < NTOK; ++c) mx = fmaxf(mx, row[c]);
    float sum = 0.f;
    for (int c = 0; c < NTOK; ++c) {
      float e = __expf(row[c] - mx);
      row[c] = e; sum += e;
    }
    float inv = 1.0f / sum;
    __bf16* prow = pr + threadIdx.x * NPAD;
    for (int c = 0; c < NTOK; ++c) prow[c] = (__bf16)(row[c] * inv);
    for (int c = NTOK; c < NPAD; ++c) prow[c] = (__bf16)0.0f;
  }
  __syncthreads();

  // O(16x64) = P(16x224) * V(224x64); waves 0..3 own one 16-col tile each,
  // B fragments come from the V^T slab in LDS (ds_load_b128 path).
  if (wave < 4) {
    v8f acc = {};
#pragma unroll
    for (int kt = 0; kt < NPAD; kt += 32) {
      v16bf a  = load_frag_rows((const __bf16*)pr, NPAD, 0, kt, lane);
      v16bf bb = load_frag_rows((const __bf16*)vt, NPAD, wave * 16, kt, lane);
      acc = wmma_bf16(a, bb, acc);
    }
    int n  = lane & 15;
    int mh = 8 * (lane >> 4);
#pragma unroll
    for (int r = 0; r < 8; ++r) {
      int tok = qt * 16 + mh + r;
      if (tok < NTOK)
        attn_out[((size_t)(b * NTOK) + tok) * D_ + h * HD_ + wave * 16 + n] =
            (__bf16)acc[r];
    }
  }
}

// bf16 cast of an f32 activation buffer
__global__ void cast_act_kernel(const float* __restrict__ s, __bf16* __restrict__ d,
                                size_t n) {
  for (size_t i = blockIdx.x * (size_t)blockDim.x + threadIdx.x; i < n;
       i += (size_t)gridDim.x * blockDim.x)
    d[i] = (__bf16)s[i];
}

// mean squared error -> atomic add into out[0] (pre-zeroed)
__global__ void mse_kernel(const float* __restrict__ a, const float* __restrict__ b,
                           float* __restrict__ out, size_t n) {
  __shared__ float red[256];
  float s = 0.f;
  for (size_t i = blockIdx.x * (size_t)blockDim.x + threadIdx.x; i < n;
       i += (size_t)gridDim.x * blockDim.x) {
    float d = a[i] - b[i];
    s += d * d;
  }
  red[threadIdx.x] = s;
  __syncthreads();
  for (int o = 128; o > 0; o >>= 1) {
    if (threadIdx.x < o) red[threadIdx.x] += red[threadIdx.x + o];
    __syncthreads();
  }
  if (threadIdx.x == 0) atomicAdd(out, red[0] / (float)n);
}

// ---------------- host side ----------------
struct EncW {
  const __bf16 *wqkv, *wo, *w1, *w2;                     // bf16 [DEPTH, ...]
  const float *ln1g, *ln1b, *bqkv, *bo, *ln2g, *ln2b, *b1, *b2;  // raw f32
};

static inline size_t al256(size_t x) { return (x + 255) & ~(size_t)255; }

static void run_encoder(float* xf, const EncW& e, __bf16* h_bf, __bf16* qkv_bf,
                        __bf16* attn_bf, __bf16* mid_bf, hipStream_t stream) {
  dim3 blk(256);
  for (int l = 0; l < DEPTH_; ++l) {
    const __bf16* Wqkv = e.wqkv + (size_t)l * QKVD * D_;
    const __bf16* Wo   = e.wo   + (size_t)l * D_ * D_;
    const __bf16* W1   = e.w1   + (size_t)l * DFF_ * D_;
    const __bf16* W2   = e.w2   + (size_t)l * D_ * DFF_;

    ln_kernel<<<MTOK, blk, 0, stream>>>(xf, e.ln1g + l * D_, e.ln1b + l * D_, h_bf);
    gemm_bf16_kernel<<<dim3(QKVD / BN, MTOK / BM), blk, 0, stream>>>(
        h_bf, Wqkv, e.bqkv + l * QKVD, nullptr, nullptr, qkv_bf,
        MTOK, QKVD, D_, 0);
    attn_kernel<<<dim3(13, H_, BATCH), blk, 0, stream>>>(qkv_bf, attn_bf);
    gemm_bf16_kernel<<<dim3(D_ / BN, MTOK / BM), blk, 0, stream>>>(
        attn_bf, Wo, e.bo + l * D_, xf, xf, nullptr, MTOK, D_, D_, 0);
    ln_kernel<<<MTOK, blk, 0, stream>>>(xf, e.ln2g + l * D_, e.ln2b + l * D_, h_bf);
    gemm_bf16_kernel<<<dim3(DFF_ / BN, MTOK / BM), blk, 0, stream>>>(
        h_bf, W1, e.b1 + l * DFF_, nullptr, nullptr, mid_bf,
        MTOK, DFF_, D_, 1 /*relu*/);
    gemm_bf16_kernel<<<dim3(D_ / BN, MTOK / BM), blk, 0, stream>>>(
        mid_bf, W2, e.b2 + l * D_, xf, xf, nullptr, MTOK, D_, DFF_, 0);
  }
}

extern "C" void kernel_launch(void* const* d_in, const int* in_sizes, int n_in,
                              void* d_out, int out_size, void* d_ws, size_t ws_size,
                              hipStream_t stream) {
  (void)in_sizes; (void)n_in; (void)out_size; (void)ws_size;
  // ---- inputs (flattened setup_inputs() dict order) ----
  const float* imgs          = (const float*)d_in[0];
  const unsigned char* cmask = (const unsigned char*)d_in[1];
  const unsigned char* tmask = (const unsigned char*)d_in[2];
  const float* conv_w        = (const float*)d_in[3];
  const float* conv_b        = (const float*)d_in[4];
  const float* pos_embed     = (const float*)d_in[5];
  const float* ctx_p[12]; const float* tgt_p[12];
  for (int i = 0; i < 12; ++i) ctx_p[i] = (const float*)d_in[6 + i];
  for (int i = 0; i < 12; ++i) tgt_p[i] = (const float*)d_in[18 + i];
  const float* pred_w1 = (const float*)d_in[30];
  const float* pred_b1 = (const float*)d_in[31];
  const float* pred_w2 = (const float*)d_in[32];
  const float* pred_b2 = (const float*)d_in[33];

  float* out_loss = (float*)d_out;
  float* out_pred = out_loss + 1;
  float* out_tgt  = out_pred + OUTEL;

  // ---- workspace bump allocator ----
  char* ws = (char*)d_ws;
  size_t off = 0;
  auto alloc = [&](size_t bytes) -> void* {
    void* p = ws + off;
    off = al256(off + bytes);
    return p;
  };
  const size_t nWqkv = (size_t)DEPTH_ * QKVD * D_;
  const size_t nWo   = (size_t)DEPTH_ * D_ * D_;
  const size_t nW1   = (size_t)DEPTH_ * DFF_ * D_;
  const size_t nW2   = (size_t)DEPTH_ * D_ * DFF_;

  __bf16* convw_bf = (__bf16*)alloc((size_t)D_ * D_ * 2);
  __bf16* enc_w[2][4];
  for (int e = 0; e < 2; ++e) {
    enc_w[e][0] = (__bf16*)alloc(nWqkv * 2);
    enc_w[e][1] = (__bf16*)alloc(nWo * 2);
    enc_w[e][2] = (__bf16*)alloc(nW1 * 2);
    enc_w[e][3] = (__bf16*)alloc(nW2 * 2);
  }
  __bf16* predw1_bf = (__bf16*)alloc((size_t)PREDD * D_ * 2);
  __bf16* predw2_bf = (__bf16*)alloc((size_t)D_ * PREDD * 2);

  __bf16* patches_bf = (__bf16*)alloc((size_t)MTOK * D_ * 2);
  float*  x_f32      = (float*)alloc((size_t)MTOK * D_ * 4);
  float*  xg_ctx     = (float*)alloc((size_t)MTOK * D_ * 4);
  float*  xg_tgt     = (float*)alloc((size_t)MTOK * D_ * 4);
  __bf16* h_bf       = (__bf16*)alloc((size_t)MTOK * D_ * 2);
  __bf16* qkv_bf     = (__bf16*)alloc((size_t)(MTOK + QKV_PADROWS) * QKVD * 2);
  __bf16* attn_bf    = (__bf16*)alloc((size_t)MTOK * D_ * 2);
  __bf16* mid_bf     = (__bf16*)alloc((size_t)MTOK * DFF_ * 2);  // also predictor mid

  dim3 blk(256);
  auto gsz = [](size_t n) { return dim3((unsigned)((n + 255) / 256)); };

  // ---- weight conversion fp32 -> bf16 ----
  cvt_f32_bf16_kernel<<<gsz((size_t)D_ * D_), blk, 0, stream>>>(conv_w, convw_bf,
                                                                (size_t)D_ * D_);
  const float* const* pp[2] = {ctx_p, tgt_p};
  for (int e = 0; e < 2; ++e) {
    cvt_f32_bf16_kernel<<<gsz(nWqkv), blk, 0, stream>>>(pp[e][2], enc_w[e][0], nWqkv);
    cvt_f32_bf16_kernel<<<gsz(nWo),   blk, 0, stream>>>(pp[e][4], enc_w[e][1], nWo);
    cvt_f32_bf16_kernel<<<gsz(nW1),   blk, 0, stream>>>(pp[e][8], enc_w[e][2], nW1);
    cvt_f32_bf16_kernel<<<gsz(nW2),   blk, 0, stream>>>(pp[e][10], enc_w[e][3], nW2);
  }
  cvt_f32_bf16_kernel<<<gsz((size_t)PREDD * D_), blk, 0, stream>>>(
      pred_w1, predw1_bf, (size_t)PREDD * D_);
  cvt_f32_bf16_kernel<<<gsz((size_t)D_ * PREDD), blk, 0, stream>>>(
      pred_w2, predw2_bf, (size_t)D_ * PREDD);

  // ---- patch embed + pos ----
  patch_extract_kernel<<<gsz((size_t)MTOK * D_), blk, 0, stream>>>(imgs, patches_bf);
  gemm_bf16_kernel<<<dim3(D_ / BN, MTOK / BM), blk, 0, stream>>>(
      patches_bf, convw_bf, conv_b, nullptr, x_f32, nullptr, MTOK, D_, D_, 0);
  add_pos_kernel<<<gsz((size_t)MTOK * D_), blk, 0, stream>>>(x_f32, pos_embed);

  // ---- gather/pad: context keeps ~cmask, target keeps tmask ----
  gather_pad_kernel<<<BATCH, blk, 0, stream>>>(x_f32, cmask, /*invert=*/1, xg_ctx);
  gather_pad_kernel<<<BATCH, blk, 0, stream>>>(x_f32, tmask, /*invert=*/0, xg_tgt);

  // ---- zero qkv pad tail once per launch (GEMMs only touch rows < MTOK) ----
  zero_bf16_kernel<<<gsz((size_t)QKV_PADROWS * QKVD), blk, 0, stream>>>(
      qkv_bf + (size_t)MTOK * QKVD, (size_t)QKV_PADROWS * QKVD);

  // ---- encoders ----
  EncW ec = {enc_w[0][0], enc_w[0][1], enc_w[0][2], enc_w[0][3],
             ctx_p[0], ctx_p[1], ctx_p[3], ctx_p[5], ctx_p[6], ctx_p[7],
             ctx_p[9], ctx_p[11]};
  EncW et = {enc_w[1][0], enc_w[1][1], enc_w[1][2], enc_w[1][3],
             tgt_p[0], tgt_p[1], tgt_p[3], tgt_p[5], tgt_p[6], tgt_p[7],
             tgt_p[9], tgt_p[11]};
  run_encoder(xg_ctx, ec, h_bf, qkv_bf, attn_bf, mid_bf, stream);
  run_encoder(xg_tgt, et, h_bf, qkv_bf, attn_bf, mid_bf, stream);

  // ---- target features straight to output ----
  copy_f32_kernel<<<gsz((size_t)OUTEL), blk, 0, stream>>>(xg_tgt, out_tgt,
                                                          (size_t)OUTEL);

  // ---- predictor: gelu(ctx @ w1^T + b1) @ w2^T + b2 ----
  cast_act_kernel<<<gsz((size_t)MTOK * D_), blk, 0, stream>>>(xg_ctx, h_bf,
                                                              (size_t)MTOK * D_);
  gemm_bf16_kernel<<<dim3(PREDD / BN, MTOK / BM), blk, 0, stream>>>(
      h_bf, predw1_bf, pred_b1, nullptr, nullptr, mid_bf, MTOK, PREDD, D_,
      2 /*gelu*/);
  gemm_bf16_kernel<<<dim3(D_ / BN, MTOK / BM), blk, 0, stream>>>(
      mid_bf, predw2_bf, pred_b2, nullptr, out_pred, nullptr, MTOK, D_, PREDD, 0);

  // ---- loss = mean((pred - tgt)^2) ----
  zero_f32_kernel<<<1, 1, 0, stream>>>(out_loss, 1);
  mse_kernel<<<1024, blk, 0, stream>>>(out_pred, out_tgt, out_loss, (size_t)OUTEL);
}
